// MultiHeadAttention_2576980377734
// MI455X (gfx1250) — matrix operand order repaired and checked by reading, not compiled, not run on hardware
//
#include <hip/hip_runtime.h>

// ---------------------------------------------------------------------------
// MultiHeadAttention forward for MI455X (gfx1250, wave32, WMMA).
// B=2, S=2048, D=1024, H=16, DK=64.  d_out = [ out (B*S*D f32) | attn (B*H*S*S f32) ]
// All matmuls use v_wmma_f32_16x16x32_bf16 (f32 accumulate); softmax in f32.
// GEMM uses 2x2 register blocking per wave (4 wmma / k-step) and stages the
// bf16 weight tile with GLOBAL_LOAD_ASYNC_TO_LDS_B128 (ASYNCcnt path).
// Workspace (48MB): bf16 WQ/WK/WV/WO, bf16 Qh/Kh, bf16 Vt, f32 ctx.
// ---------------------------------------------------------------------------

#define BB 2
#define SS 2048
#define DD 1024
#define HH 16
#define DKK 64

typedef __attribute__((ext_vector_type(16))) __bf16       v16bf;
typedef __attribute__((ext_vector_type(8)))  float        v8f;
typedef __attribute__((ext_vector_type(8)))  unsigned int v8u;

// ---- bf16 helpers (round-to-nearest-even) ---------------------------------
static __device__ __forceinline__ unsigned short f2bf(float f) {
  unsigned int u = __builtin_bit_cast(unsigned int, f);
  u += 0x7FFFu + ((u >> 16) & 1u);
  return (unsigned short)(u >> 16);
}
static __device__ __forceinline__ unsigned int pack2bf(float lo, float hi) {
  return (unsigned int)f2bf(lo) | ((unsigned int)f2bf(hi) << 16);
}

// ---- WMMA fragment gathers, exact CDNA5 VGPR layouts (ISA 7.12.2) ---------
static __device__ __forceinline__ v16bf frag_a(const unsigned short* base,
                                               int stride, int lane, int koff) {
  const int m = lane & 15, hh = (lane >> 4) & 1;
  v8u u;
#pragma unroll
  for (int i = 0; i < 8; ++i) {
    const int k = koff + ((i & 4) << 2) + ((i & 3) << 1) + (hh << 3);
    u[i] = *(const unsigned int*)(base + m * stride + k);
  }
  return __builtin_bit_cast(v16bf, u);
}
static __device__ __forceinline__ v16bf frag_b(const unsigned short* base,
                                               int stride, int lane, int koff) {
  const int n = lane & 15, hh = (lane >> 4) & 1;
  v8u u;
#pragma unroll
  for (int i = 0; i < 8; ++i) {
    const int k = koff + ((i & 4) << 2) + ((i & 3) << 1) + (hh << 3);
    u[i] = *(const unsigned int*)(base + n * stride + k);
  }
  return __builtin_bit_cast(v16bf, u);
}
static __device__ __forceinline__ v8f wmma_bf16(v16bf a, v16bf b, v8f c) {
  return __builtin_amdgcn_wmma_f32_16x16x32_bf16(false, a, false, b,
                                                 (short)0, c, false, false);
}

// ---- gfx1250 async global->LDS copy (ASYNCcnt-tracked, no VGPR bounce) ----
static __device__ __forceinline__ void async_copy_b128(unsigned short* lds_dst,
                                                       const unsigned short* gsrc) {
  // Low 32 bits of the flat shared pointer == LDS byte offset (ISA 10.2).
  unsigned int ldsa = (unsigned int)(unsigned long long)lds_dst;
  unsigned long long ga = (unsigned long long)gsrc;
  asm volatile("global_load_async_to_lds_b128 %0, %1, off"
               :: "v"(ldsa), "v"(ga) : "memory");
}
static __device__ __forceinline__ void wait_async0() {
  asm volatile("s_wait_asynccnt 0" ::: "memory");
}

// ---- wave32 reductions ----------------------------------------------------
static __device__ __forceinline__ float wave_max(float v) {
#pragma unroll
  for (int o = 16; o > 0; o >>= 1) v = fmaxf(v, __shfl_xor(v, o, 32));
  return v;
}
static __device__ __forceinline__ float wave_sum(float v) {
#pragma unroll
  for (int o = 16; o > 0; o >>= 1) v += __shfl_xor(v, o, 32);
  return v;
}

// ===========================================================================
// Kernel 1: f32 -> bf16 conversion (weights)
// ===========================================================================
__global__ void k_f32_to_bf16(const float* __restrict__ in,
                              unsigned short* __restrict__ out, int n) {
  int i = blockIdx.x * blockDim.x + threadIdx.x;
  if (i < n) out[i] = f2bf(in[i]);
}

// ===========================================================================
// Kernel 2: GEMM  C[M,N] = A[M,K](f32) * W[N,K](bf16)^T + bias
// block 256 (8 waves, 2x4), block tile 64(M) x 128(N), K staged in 32-steps.
// Each wave computes 32x32 (2x2 WMMA tiles) -> 4 wmma per k-step.
// W tile staged via global_load_async_to_lds_b128; A tile converted in-flight.
//   MODE 0: store bf16 head-split      Qh/Kh[((b*H+h)*S+s)*DK+dk]
//   MODE 1: store bf16 head-transposed Vt[((b*H+h)*DK+dk)*S+s]
//   MODE 2: store f32 row-major        out[row*N+col]
// ===========================================================================
template <int MODE>
__global__ void k_gemm(const float* __restrict__ A,
                       const unsigned short* __restrict__ W,
                       const float* __restrict__ bias, void* __restrict__ Out) {
  constexpr int K = DD, N = DD;
  __shared__ unsigned short Alds[64 * 32];    // 4KB
  __shared__ unsigned short Wlds[128 * 32];   // 8KB

  const int tid = threadIdx.x, lane = tid & 31, wave = tid >> 5;
  const int wm = wave >> 2, wn = wave & 3;    // 2 x 4 wave grid
  const int m0 = blockIdx.x * 64, n0 = blockIdx.y * 128;

  v8f acc[2][2] = {{{}, {}}, {{}, {}}};

  const int ar = tid >> 2, ac = (tid & 3) << 3;  // A stage: 64 rows x 32 cols, 8/thread

  for (int k0 = 0; k0 < K; k0 += 32) {
    // W tile: 128x32 bf16 copy via async path (2 x b128 per thread)
#pragma unroll
    for (int c = 0; c < 2; ++c) {
      const int ch = tid * 2 + c;            // 512 chunks of 8 ushorts
      const int rw = ch >> 2, wc8 = (ch & 3) << 3;
      async_copy_b128(&Wlds[rw * 32 + wc8],
                      W + (size_t)(n0 + rw) * K + k0 + wc8);
    }
    // A tile: f32 -> bf16 convert while staging
    {
      const float4 f0 = *(const float4*)(A + (size_t)(m0 + ar) * K + k0 + ac);
      const float4 f1 = *(const float4*)(A + (size_t)(m0 + ar) * K + k0 + ac + 4);
      uint4 pk;
      pk.x = pack2bf(f0.x, f0.y);
      pk.y = pack2bf(f0.z, f0.w);
      pk.z = pack2bf(f1.x, f1.y);
      pk.w = pack2bf(f1.z, f1.w);
      *(uint4*)(&Alds[ar * 32 + ac]) = pk;
    }
    if (k0 + 32 < K)
      __builtin_prefetch(A + (size_t)(m0 + ar) * K + k0 + 32 + ac, 0, 1);
    wait_async0();
    __syncthreads();

    const v16bf a0 = frag_a(Alds + (wm * 32) * 32, 32, lane, 0);
    const v16bf a1 = frag_a(Alds + (wm * 32 + 16) * 32, 32, lane, 0);
    const v16bf b0 = frag_b(Wlds + (wn * 32) * 32, 32, lane, 0);
    const v16bf b1 = frag_b(Wlds + (wn * 32 + 16) * 32, 32, lane, 0);
    acc[0][0] = wmma_bf16(a0, b0, acc[0][0]);
    acc[0][1] = wmma_bf16(a0, b1, acc[0][1]);
    acc[1][0] = wmma_bf16(a1, b0, acc[1][0]);
    acc[1][1] = wmma_bf16(a1, b1, acc[1][1]);
    __syncthreads();
  }

  // epilogue: tile (i,j), element (m = r + 8*(lane>>4), n = lane&15) in acc[r]
#pragma unroll
  for (int i = 0; i < 2; ++i) {
#pragma unroll
    for (int j = 0; j < 2; ++j) {
      const int col = n0 + wn * 32 + j * 16 + (lane & 15);
      const float bv = bias[col];
#pragma unroll
      for (int r = 0; r < 8; ++r) {
        const int row = m0 + wm * 32 + i * 16 + r + ((lane >> 4) << 3);
        const float v = acc[i][j][r] + bv;
        if (MODE == 2) {
          ((float*)Out)[(size_t)row * N + col] = v;
        } else {
          const int b = row >> 11, s = row & (SS - 1);
          const int h = col >> 6, dk = col & (DKK - 1);
          if (MODE == 0)
            ((unsigned short*)Out)[(size_t)((b * HH + h) * SS + s) * DKK + dk] = f2bf(v);
          else
            ((unsigned short*)Out)[(size_t)((b * HH + h) * DKK + dk) * SS + s] = f2bf(v);
        }
      }
    }
  }
}

// ===========================================================================
// Kernel 3: scores = (Qh . Kh^T)/8, causal mask, softmax -> attn (f32, d_out)
// grid (S/16, H, B), block 256 (8 waves). LDS: 16x2048 f32 scores + Q tile.
// Causal: only key tiles t <= qtile are computed (WMMA loop is wave-uniform).
// ===========================================================================
__global__ void k_scores_softmax(const unsigned short* __restrict__ Qh,
                                 const unsigned short* __restrict__ Kh,
                                 float* __restrict__ attn) {
  extern __shared__ char smem[];
  float* sc = (float*)smem;                                    // 16 x 2048 f32
  unsigned short* Qs = (unsigned short*)(smem + 16 * SS * 4);  // 16 x 64 bf16

  const int tid = threadIdx.x, lane = tid & 31, wave = tid >> 5;
  const int qtile = blockIdx.x, q0 = qtile * 16;
  const int h = blockIdx.y, b = blockIdx.z;
  const size_t headQ = (size_t)(b * HH + h) * SS * DKK;

  {  // stage Q tile 16x64
    const int r = tid >> 4, c = (tid & 15) << 2;
    *(uint2*)(&Qs[r * DKK + c]) =
        *(const uint2*)(Qh + headQ + (size_t)(q0 + r) * DKK + c);
  }
  __syncthreads();

  const unsigned short* Kbase = Kh + headQ;
  for (int t = wave; t <= qtile; t += 8) {  // uniform per wave
    v8f acc = {};
#pragma unroll
    for (int c = 0; c < 2; ++c) {           // DK=64 -> two K=32 steps
      v16bf a = frag_a(Qs, DKK, lane, c * 32);
      v16bf bf = frag_b(Kbase + (size_t)t * 16 * DKK, DKK, lane, c * 32);
      acc = wmma_bf16(a, bf, acc);
    }
    const int n = lane & 15, hh = lane >> 4;
#pragma unroll
    for (int r = 0; r < 8; ++r)
      sc[(r + (hh << 3)) * SS + t * 16 + n] = acc[r] * 0.125f;  // 1/sqrt(64)
  }
  __syncthreads();

  for (int r = wave; r < 16; r += 8) {  // softmax, 2 rows per wave
    const int qi = q0 + r;
    float* row = sc + r * SS;
    float mx = -3.0e38f;
    for (int j = lane; j <= qi; j += 32) mx = fmaxf(mx, row[j]);
    mx = wave_max(mx);
    float sum = 0.f;
    for (int j = lane; j <= qi; j += 32) {
      const float e = __expf(row[j] - mx);
      row[j] = e;
      sum += e;
    }
    sum = wave_sum(sum);
    const float inv = 1.0f / sum;
    float* orow = attn + ((size_t)(b * HH + h) * SS + qi) * SS;
    for (int j = lane; j < SS; j += 32) orow[j] = (j <= qi) ? row[j] * inv : 0.0f;
  }
}

// ===========================================================================
// Kernel 4: ctx = attn @ Vh  (per head GEMM M=S, N=DK=64, K=S, causal-trimmed)
// grid (S/16, H, B), block 128 (4 waves, one 16x16 dk tile each).
// attn f32 read from d_out, converted to bf16 while staging to LDS.
// Vt is [b,h,dk,s] -> already B-operand (N x K) layout.
// ===========================================================================
__global__ void k_ctx(const float* __restrict__ attn,
                      const unsigned short* __restrict__ Vt,
                      float* __restrict__ ctx) {
  __shared__ unsigned short Aatt[16 * 32];
  const int tid = threadIdx.x, lane = tid & 31, wave = tid >> 5;
  const int qtile = blockIdx.x, q0 = qtile * 16;
  const int h = blockIdx.y, b = blockIdx.z;
  const int n0 = wave * 16;  // dk tile

  const float* arow = attn + ((size_t)(b * HH + h) * SS + q0) * SS;
  const unsigned short* Vb = Vt + (size_t)(b * HH + h) * DKK * SS;

  v8f acc = {};
  const int sr = tid >> 3, sc = (tid & 7) << 2;  // stage 16 x 32
  for (int kt = 0; kt <= qtile; ++kt) {          // causal trim, uniform
    const float4 f = *(const float4*)(arow + (size_t)sr * SS + kt * 32 + sc);
    *(unsigned int*)(&Aatt[sr * 32 + sc])     = pack2bf(f.x, f.y);
    *(unsigned int*)(&Aatt[sr * 32 + sc + 2]) = pack2bf(f.z, f.w);
    __syncthreads();
    v16bf a = frag_a(Aatt, 32, lane, 0);
    v16bf bf = frag_b(Vb + (size_t)n0 * SS, SS, lane, kt * 32);
    acc = wmma_bf16(a, bf, acc);
    __syncthreads();
  }

  const int dk = n0 + (lane & 15);
#pragma unroll
  for (int r = 0; r < 8; ++r) {
    const int m = r + ((lane >> 4) << 3);
    ctx[((size_t)(b * SS + q0 + m)) * DD + h * DKK + dk] = acc[r];
  }
}

// ===========================================================================
extern "C" void kernel_launch(void* const* d_in, const int* in_sizes, int n_in,
                              void* d_out, int out_size, void* d_ws, size_t ws_size,
                              hipStream_t stream) {
  const float* Q  = (const float*)d_in[0];
  const float* K  = (const float*)d_in[1];
  const float* V  = (const float*)d_in[2];
  // d_in[3] = mask (causal tril) handled analytically
  const float* WQ = (const float*)d_in[4];
  const float* bQ = (const float*)d_in[5];
  const float* WK = (const float*)d_in[6];
  const float* bK = (const float*)d_in[7];
  const float* WV = (const float*)d_in[8];
  const float* bV = (const float*)d_in[9];
  const float* WO = (const float*)d_in[10];
  const float* bO = (const float*)d_in[11];

  char* ws = (char*)d_ws;
  const size_t WBYTES = (size_t)DD * DD * 2;             // 2MB per bf16 weight
  const size_t HBYTES = (size_t)BB * HH * SS * DKK * 2;  // 8MB per bf16 head tensor
  unsigned short* Wqb = (unsigned short*)(ws);
  unsigned short* Wkb = (unsigned short*)(ws + WBYTES);
  unsigned short* Wvb = (unsigned short*)(ws + 2 * WBYTES);
  unsigned short* Wob = (unsigned short*)(ws + 3 * WBYTES);
  unsigned short* Qh  = (unsigned short*)(ws + 4 * WBYTES);
  unsigned short* Kh  = (unsigned short*)(ws + 4 * WBYTES + HBYTES);
  unsigned short* Vt  = (unsigned short*)(ws + 4 * WBYTES + 2 * HBYTES);
  float*          Ctx = (float*)(ws + 4 * WBYTES + 3 * HBYTES);  // 16MB f32

  float* out  = (float*)d_out;
  float* attn = out + (size_t)BB * SS * DD;

  // 1) weights f32 -> bf16
  const int wn = DD * DD;
  k_f32_to_bf16<<<(wn + 255) / 256, 256, 0, stream>>>(WQ, Wqb, wn);
  k_f32_to_bf16<<<(wn + 255) / 256, 256, 0, stream>>>(WK, Wkb, wn);
  k_f32_to_bf16<<<(wn + 255) / 256, 256, 0, stream>>>(WV, Wvb, wn);
  k_f32_to_bf16<<<(wn + 255) / 256, 256, 0, stream>>>(WO, Wob, wn);

  // 2) projections
  dim3 gg(BB * SS / 64, DD / 128);
  k_gemm<0><<<gg, 256, 0, stream>>>(Q, Wqb, bQ, (void*)Qh);
  k_gemm<0><<<gg, 256, 0, stream>>>(K, Wkb, bK, (void*)Kh);
  k_gemm<1><<<gg, 256, 0, stream>>>(V, Wvb, bV, (void*)Vt);

  // 3) scores + causal softmax -> attn (f32, second output)
  const size_t smem = (size_t)16 * SS * 4 + (size_t)16 * DKK * 2;  // 130KB
  k_scores_softmax<<<dim3(SS / 16, HH, BB), 256, smem, stream>>>(Qh, Kh, attn);

  // 4) ctx = attn @ V
  k_ctx<<<dim3(SS / 16, HH, BB), 128, 0, stream>>>(attn, Vt, Ctx);

  // 5) out = ctx @ WO^T + bO  (f32, first output)
  k_gemm<2><<<gg, 256, 0, stream>>>(Ctx, Wob, bO, d_out);
}
